// RetNet_4114578670500
// MI455X (gfx1250) — matrix operand order repaired and checked by reading, not compiled, not run on hardware
//
#include <hip/hip_runtime.h>

#define DEV __device__ __forceinline__

typedef __attribute__((ext_vector_type(8)))  unsigned short us8;
typedef __attribute__((ext_vector_type(16))) __bf16         v16bf;
typedef __attribute__((ext_vector_type(8)))  float          v8f;

struct FragF  { union { v8f   v; float f[8]; }; };
struct FragBF { union { v16bf v; us8 h[2]; unsigned short u[16]; }; };

DEV unsigned short f2bf(float x){
  union { float f; unsigned u; } c; c.f = x;
  unsigned r = (c.u + 0x7FFFu + ((c.u >> 16) & 1u)) >> 16;
  return (unsigned short)r;
}
DEV float bf2f(unsigned short b){
  union { unsigned u; float f; } c; c.u = ((unsigned)b) << 16;
  return c.f;
}

// D = A(16x32 bf16) * B(32x16 bf16) + C(16x16 f32)
DEV v8f wmma_bf16(v16bf a, v16bf b, v8f c){
  return __builtin_amdgcn_wmma_f32_16x16x32_bf16(
      /*neg_a=*/false, a, /*neg_b=*/false, b,
      /*c_mod=*/(short)0, c, /*reuse_a=*/false, /*reuse_b=*/false);
}

// CDNA5 async copy: 16B global -> LDS, tracked by ASYNCcnt.
// LDS dest VGPR = low 32 bits of the generic pointer (ISA 10.2: LDS aperture
// uses addr[31:0]).
DEV void async_b128(const void* g, void* l){
  unsigned lds = (unsigned)(unsigned long long)l;
  unsigned long long ga = (unsigned long long)g;
  asm volatile("global_load_async_to_lds_b128 %0, %1, off"
               :: "v"(lds), "v"(ga) : "memory");
}
#define WAIT_ASYNC(n) asm volatile("s_wait_asynccnt " #n ::: "memory")

// A-fragment (16x32, row-major source): lane m=l&15, hf=l>>4
//   e in 0..7  -> k = hf*8 + e ;  e in 8..15 -> k = 16 + hf*8 + (e-8)
DEV v16bf load_afrag(const unsigned short* base, int stride, int lane){
  int m = lane & 15, hf = lane >> 4;
  FragBF a;
  a.h[0] = *(const us8*)(base + m * stride + hf * 8);
  a.h[1] = *(const us8*)(base + m * stride + 16 + hf * 8);
  return a.v;
}
// B-fragment (32x16, source stored N-major: row n holds k=0..31 contiguous)
//   lane n=l&15, hf=l>>4 : e in 0..7 -> k = hf*8 + e ; e in 8..15 -> k = 16 + hf*8 + (e-8)
DEV v16bf load_bfrag(const unsigned short* base, int stride, int lane){
  int n = lane & 15, hf = lane >> 4;
  FragBF b;
  b.h[0] = *(const us8*)(base + n * stride + hf * 8);
  b.h[1] = *(const us8*)(base + n * stride + 16 + hf * 8);
  return b.v;
}

// ---------------- weight convert + transpose: Wt[n*K+k] = bf16(W[k*N+n]) ----
__global__ void cvt_t_kernel(const float* __restrict__ W,
                             unsigned short* __restrict__ Wt, int K, int N){
  int i = blockIdx.x * 256 + threadIdx.x;
  if (i >= K * N) return;
  int k = i / N, n = i - k * N;
  Wt[(size_t)n * K + k] = f2bf(W[i]);
}

// ---------------- layernorm over D=1024, fp32 in -> bf16 out ---------------
__global__ void __launch_bounds__(256) ln_kernel(const float* __restrict__ X,
                                                 unsigned short* __restrict__ O,
                                                 const float* __restrict__ sc,
                                                 const float* __restrict__ bi){
  int row = blockIdx.x, t = threadIdx.x;
  const float* x = X + (size_t)row * 1024 + t * 4;
  float4 v = *(const float4*)x;
  float s = v.x + v.y + v.z + v.w;
  float q = v.x*v.x + v.y*v.y + v.z*v.z + v.w*v.w;
  __shared__ float ss[256], sq[256];
  ss[t] = s; sq[t] = q; __syncthreads();
  for (int off = 128; off > 0; off >>= 1){
    if (t < off){ ss[t] += ss[t+off]; sq[t] += sq[t+off]; }
    __syncthreads();
  }
  float mu  = ss[0] * (1.f/1024.f);
  float var = sq[0] * (1.f/1024.f) - mu * mu;
  float inv = rsqrtf(var + 1e-5f);
  unsigned short* op = O + (size_t)row * 1024 + t * 4;
  float a[4] = {v.x, v.y, v.z, v.w};
  #pragma unroll
  for (int i = 0; i < 4; ++i){
    int d = t*4 + i;
    op[i] = f2bf((a[i] - mu) * inv * sc[d] + bi[d]);
  }
}

// ---------------- rotary on Q (+sin) and K (-sin), fp32 -> bf16 ------------
__global__ void rotary_kernel(const float* __restrict__ Qf, const float* __restrict__ Kf,
                              unsigned short* __restrict__ Qh, unsigned short* __restrict__ Kh){
  int p  = blockIdx.x * 256 + threadIdx.x;  // pair over B*S*H*64
  int j  = p & 63;
  int h  = (p >> 6) & 7;
  int bs = p >> 9;
  int s  = bs & 1023;
  size_t idx = ((size_t)bs * 8 + h) * 128 + 2*j;
  // inv_freq = 10000^(-2j/128) = exp(-j * ln(10000)/64)
  float ang = (float)s * __expf(-(float)j * 0.14391156f);
  float cv, sv; __sincosf(ang, &sv, &cv);
  float q1 = Qf[idx], q2 = Qf[idx+1];
  Qh[idx]   = f2bf(q1*cv - q2*sv);
  Qh[idx+1] = f2bf(q1*sv + q2*cv);
  float k1 = Kf[idx], k2 = Kf[idx+1];
  Kh[idx]   = f2bf(k1*cv + k2*sv);
  Kh[idx+1] = f2bf(-k1*sv + k2*cv);
}

// ---------------- groupnorm(DH=128) * gn + bias, gated by swish-G ----------
__global__ void __launch_bounds__(256) gate_gn_kernel(const float* __restrict__ Y,
                               const unsigned short* __restrict__ G,
                               const float* __restrict__ gs, const float* __restrict__ gb,
                               unsigned short* __restrict__ GY){
  int row = blockIdx.x;                         // b*S + s
  int h = threadIdx.x >> 5, lane = threadIdx.x & 31;
  size_t base = (size_t)row * 1024 + h * 128 + lane * 4;
  float4 v = *(const float4*)(Y + base);
  float s = v.x + v.y + v.z + v.w;
  float q = v.x*v.x + v.y*v.y + v.z*v.z + v.w*v.w;
  for (int m = 16; m >= 1; m >>= 1){ s += __shfl_xor(s, m, 32); q += __shfl_xor(q, m, 32); }
  float mu  = s * (1.f/128.f);
  float var = q * (1.f/128.f) - mu * mu;
  float inv = rsqrtf(var + 1e-5f);
  float a[4] = {v.x, v.y, v.z, v.w};
  #pragma unroll
  for (int i = 0; i < 4; ++i){
    int d = h*128 + lane*4 + i;
    float val = (a[i] - mu) * inv * gs[d] + gb[d];
    GY[base + i] = f2bf(val * bf2f(G[base + i]));
  }
}

// ---------------- bf16 WMMA GEMM, 128x128 tile, async-LDS pipeline ---------
// A: [M,K] bf16 row-major. Bt: [N,K] bf16 (i.e. B transposed). C: fp32 acc.
// epi: 0=f32  1=f32+res  2=swish->bf16  3=gelu->bf16  4=bf16
__global__ void __launch_bounds__(256) gemm_bf16_kernel(
    const unsigned short* __restrict__ A, const unsigned short* __restrict__ Bt,
    int M, int N, int K, float* __restrict__ outF, unsigned short* __restrict__ outH,
    const float* __restrict__ res, int epi){
  __shared__ __attribute__((aligned(16))) unsigned short sA[2][128][40];
  __shared__ __attribute__((aligned(16))) unsigned short sB[2][128][40];
  int t = threadIdx.x, w = t >> 5, lane = t & 31;
  int hf = lane >> 4, ln15 = lane & 15;
  int mblk = blockIdx.y, nblk = blockIdx.x;
  int wm = w >> 1, wn = w & 1;

  int arow = t >> 1, koff = (t & 1) * 16;
  const unsigned short* gA = A  + ((size_t)(mblk*128 + arow)) * K + koff;
  const unsigned short* gB = Bt + ((size_t)(nblk*128 + arow)) * K + koff;

  // async stage of one 128x32 K-slice of A and Bt into LDS buffer `buf`
  auto stage = [&](int buf, int ks){
    const unsigned short* pA = gA + (size_t)ks * 32;
    const unsigned short* pB = gB + (size_t)ks * 32;
    async_b128(pA,     &sA[buf][arow][koff]);
    async_b128(pA + 8, &sA[buf][arow][koff + 8]);
    async_b128(pB,     &sB[buf][arow][koff]);
    async_b128(pB + 8, &sB[buf][arow][koff + 8]);
  };

  stage(0, 0);

  FragF c[8];
  #pragma unroll
  for (int i = 0; i < 8; ++i)
    #pragma unroll
    for (int r = 0; r < 8; ++r) c[i].f[r] = 0.f;

  int nk = K >> 5;
  for (int ks = 0; ks < nk; ++ks){
    int cur = ks & 1;
    if (ks + 1 < nk){
      stage(cur ^ 1, ks + 1);           // 4 more async ops in flight (8 total)
      WAIT_ASYNC(4);                    // oldest 4 (= current buffer) done
    } else {
      WAIT_ASYNC(0);
    }
    if (ks + 2 < nk){
      __builtin_prefetch(gA + (size_t)(ks+2) * 32, 0, 1);
      __builtin_prefetch(gB + (size_t)(ks+2) * 32, 0, 1);
    }
    __syncthreads();                    // everyone's slice landed

    v16bf af0 = load_afrag(&sA[cur][wm*32][0],      40, lane);
    v16bf af1 = load_afrag(&sA[cur][wm*32 + 16][0], 40, lane);
    v16bf bfr[4];
    #pragma unroll
    for (int j = 0; j < 4; ++j)
      bfr[j] = load_bfrag(&sB[cur][wn*64 + j*16][0], 40, lane);
    #pragma unroll
    for (int j = 0; j < 4; ++j){
      c[j].v     = wmma_bf16(af0, bfr[j], c[j].v);
      c[4 + j].v = wmma_bf16(af1, bfr[j], c[4 + j].v);
    }
    __syncthreads();                    // done reading before next overwrite
  }

  size_t row0 = (size_t)mblk*128 + wm*32;
  int col0 = nblk*128 + wn*64;
  #pragma unroll
  for (int i = 0; i < 8; ++i){
    int tm = i >> 2, tn = i & 3;
    #pragma unroll
    for (int r = 0; r < 8; ++r){
      size_t row = row0 + tm*16 + r + 8*hf;
      int col = col0 + tn*16 + ln15;
      size_t idx = row * (size_t)N + col;
      float v = c[i].f[r];
      if (epi == 0){
        outF[idx] = v;
      } else if (epi == 1){
        outF[idx] = v + res[idx];
      } else if (epi == 2){                    // swish
        outH[idx] = f2bf(v / (1.f + __expf(-v)));
      } else if (epi == 3){                    // gelu (tanh approx)
        float g = 0.5f * v * (1.f + tanhf(0.79788456f * (v + 0.044715f * v * v * v)));
        outH[idx] = f2bf(g);
      } else {
        outH[idx] = f2bf(v);
      }
    }
  }
}

// ---------------- fused causal retention: Y = (QK^T ⊙ decay) V -------------
// Layouts: Qh/Kh/Vh bf16 [B,S,H,128]; Y fp32 [B,S,H,128].
// Block = (query-128-tile, h, b); 8 waves * 16 query rows each.
__global__ void __launch_bounds__(256) retention_kernel(
    const unsigned short* __restrict__ Qh, const unsigned short* __restrict__ Kh,
    const unsigned short* __restrict__ Vh, float* __restrict__ Y){
  const int S = 1024;
  int nblk = blockIdx.x, h = blockIdx.y, b = blockIdx.z;
  int t = threadIdx.x, w = t >> 5, lane = t & 31;
  int hf = lane >> 4, ln15 = lane & 15;
  int nbase = nblk * 128;

  __shared__ __attribute__((aligned(16))) unsigned short sK[32][136];
  __shared__ __attribute__((aligned(16))) unsigned short sVt[128][40];
  __shared__ __attribute__((aligned(16))) unsigned short sAtt[8][16][32];

  // gamma_h = 1 - exp(linspace(ln(1/32), ln(1/512), 8)); lg = ln(gamma_h)
  float lg = __logf(1.f - __expf(-3.4657359f + (float)h * -0.39608410f));

  // Q fragments: 16 rows x 128 (K-dim) -> 4 bf16 A-frags held in registers
  int qrow = nbase + w*16 + ln15;
  const unsigned short* qp = Qh + (((size_t)b * S + qrow) * 8 + h) * 128;
  FragBF qa[4];
  #pragma unroll
  for (int kt = 0; kt < 4; ++kt){
    qa[kt].h[0] = *(const us8*)(qp + kt*32 + hf*8);
    qa[kt].h[1] = *(const us8*)(qp + kt*32 + 16 + hf*8);
  }

  FragF yacc[8];
  #pragma unroll
  for (int dt = 0; dt < 8; ++dt)
    #pragma unroll
    for (int r = 0; r < 8; ++r) yacc[dt].f[r] = 0.f;

  int msteps = (nbase + 128) >> 5;        // causal: keys up to end of this tile
  for (int mt = 0; mt < msteps; ++mt){
    int mbase = mt * 32;
    {   // stage K tile [32][128] async; V tile transposed [128][32] explicit
      int row = t >> 3, off = (t & 7) * 16;
      const unsigned short* kp = Kh + (((size_t)b * S + mbase + row) * 8 + h) * 128 + off;
      const unsigned short* vp = Vh + (((size_t)b * S + mbase + row) * 8 + h) * 128 + off;
      async_b128(kp,     &sK[row][off]);
      async_b128(kp + 8, &sK[row][off + 8]);
      us8 v0 = *(const us8*)vp, v1 = *(const us8*)(vp + 8);
      #pragma unroll
      for (int e = 0; e < 8; ++e){
        sVt[off + e][row]     = v0[e];
        sVt[off + 8 + e][row] = v1[e];
      }
      WAIT_ASYNC(0);
    }
    __syncthreads();

    // att[16x32] = Q(16x128) * K^T : K-tile rows are exactly Bt layout
    FragF c0, c1;
    #pragma unroll
    for (int r = 0; r < 8; ++r){ c0.f[r] = 0.f; c1.f[r] = 0.f; }
    #pragma unroll
    for (int kt = 0; kt < 4; ++kt){
      v16bf b0 = load_bfrag((const unsigned short*)sK + kt*32,            136, lane);
      v16bf b1 = load_bfrag((const unsigned short*)sK + 16*136 + kt*32,   136, lane);
      c0.v = wmma_bf16(qa[kt].v, b0, c0.v);
      c1.v = wmma_bf16(qa[kt].v, b1, c1.v);
    }
    // decay + causal mask, re-layout C->A through wave-private LDS
    #pragma unroll
    for (int r = 0; r < 8; ++r){
      int n  = nbase + w*16 + r + 8*hf;
      int m0 = mbase + ln15, m1 = m0 + 16;
      float a0 = (n >= m0) ? c0.f[r] * __expf(lg * (float)(n - m0)) : 0.f;
      float a1 = (n >= m1) ? c1.f[r] * __expf(lg * (float)(n - m1)) : 0.f;
      sAtt[w][r + 8*hf][ln15]      = f2bf(a0);
      sAtt[w][r + 8*hf][16 + ln15] = f2bf(a1);
    }
    v16bf aatt = load_afrag(&sAtt[w][0][0], 32, lane);   // same-wave DS: in order

    // Y(16x128) += att(16x32) * V(32x128)  (V staged transposed -> Bt layout)
    #pragma unroll
    for (int dt = 0; dt < 8; ++dt){
      v16bf vb = load_bfrag(&sVt[dt*16][0], 40, lane);
      yacc[dt].v = wmma_bf16(aatt, vb, yacc[dt].v);
    }
    __syncthreads();
  }

  #pragma unroll
  for (int dt = 0; dt < 8; ++dt)
    #pragma unroll
    for (int r = 0; r < 8; ++r){
      int n = nbase + w*16 + r + 8*hf;
      int d = dt*16 + ln15;
      Y[(((size_t)b * S + n) * 8 + h) * 128 + d] = yacc[dt].f[r];
    }
}

// ---------------------------------------------------------------------------
extern "C" void kernel_launch(void* const* d_in, const int* in_sizes, int n_in,
                              void* d_out, int out_size, void* d_ws, size_t ws_size,
                              hipStream_t stream){
  (void)in_sizes; (void)n_in; (void)out_size; (void)ws_size;
  const int L = 4, D = 1024, F = 2048, B = 4, S = 1024;
  const int BS = B * S;   // 4096

  const float* X   = (const float*)d_in[0];
  const float* Wq  = (const float*)d_in[1];
  const float* Wk  = (const float*)d_in[2];
  const float* Wv  = (const float*)d_in[3];
  const float* Wg  = (const float*)d_in[4];
  const float* Wo  = (const float*)d_in[5];
  const float* W1  = (const float*)d_in[6];
  const float* W2  = (const float*)d_in[7];
  const float* lns = (const float*)d_in[8];
  const float* lnb = (const float*)d_in[9];
  const float* gns = (const float*)d_in[10];
  const float* gnb = (const float*)d_in[11];
  float* out = (float*)d_out;

  char* p = (char*)d_ws;
  auto alloc = [&](size_t bytes) -> char* {
    char* r = p; p += (bytes + 255) & ~(size_t)255; return r;
  };
  const size_t DD = (size_t)D * D, DF = (size_t)D * F;
  unsigned short* WqT = (unsigned short*)alloc(L * DD * 2);
  unsigned short* WkT = (unsigned short*)alloc(L * DD * 2);
  unsigned short* WvT = (unsigned short*)alloc(L * DD * 2);
  unsigned short* WgT = (unsigned short*)alloc(L * DD * 2);
  unsigned short* WoT = (unsigned short*)alloc(L * DD * 2);
  unsigned short* W1T = (unsigned short*)alloc(L * DF * 2);
  unsigned short* W2T = (unsigned short*)alloc(L * DF * 2);
  unsigned short* Xn  = (unsigned short*)alloc((size_t)BS * D * 2);
  float*          Qf  = (float*)alloc((size_t)BS * D * 4);
  float*          Kf  = (float*)alloc((size_t)BS * D * 4);
  unsigned short* Qb  = (unsigned short*)alloc((size_t)BS * D * 2);
  unsigned short* Kb  = (unsigned short*)alloc((size_t)BS * D * 2);
  unsigned short* Vb  = (unsigned short*)alloc((size_t)BS * D * 2);
  unsigned short* Gb  = (unsigned short*)alloc((size_t)BS * D * 2);
  float*          Yw  = (float*)alloc((size_t)BS * D * 4);
  unsigned short* GY  = (unsigned short*)alloc((size_t)BS * D * 2);
  float*          Yr  = (float*)alloc((size_t)BS * D * 4);
  unsigned short* H1  = (unsigned short*)alloc((size_t)BS * F * 2);

  auto cvt = [&](const float* Wp, unsigned short* Wt, int Kd, int Nd){
    int total = Kd * Nd;
    cvt_t_kernel<<<dim3((total + 255) / 256), 256, 0, stream>>>(Wp, Wt, Kd, Nd);
  };
  for (int l = 0; l < L; ++l){
    cvt(Wq + l * DD, WqT + l * DD, D, D);
    cvt(Wk + l * DD, WkT + l * DD, D, D);
    cvt(Wv + l * DD, WvT + l * DD, D, D);
    cvt(Wg + l * DD, WgT + l * DD, D, D);
    cvt(Wo + l * DD, WoT + l * DD, D, D);
    cvt(W1 + l * DF, W1T + l * DF, D, F);
    cvt(W2 + l * DF, W2T + l * DF, F, D);
  }

  auto gemm = [&](const unsigned short* Ap, const unsigned short* Bp,
                  int M, int N, int Kd, float* oF, unsigned short* oH,
                  const float* res, int epi){
    gemm_bf16_kernel<<<dim3(N / 128, M / 128), 256, 0, stream>>>(
        Ap, Bp, M, N, Kd, oF, oH, res, epi);
  };

  for (int l = 0; l < L; ++l){
    const unsigned short* wq = WqT + l * DD;
    const unsigned short* wk = WkT + l * DD;
    const unsigned short* wv = WvT + l * DD;
    const unsigned short* wg = WgT + l * DD;
    const unsigned short* wo = WoT + l * DD;
    const unsigned short* w1 = W1T + l * DF;
    const unsigned short* w2 = W2T + l * DF;
    const float* Xcur = (l == 0) ? X : (const float*)out;

    ln_kernel<<<BS, 256, 0, stream>>>(Xcur, Xn, lns, lnb);
    gemm(Xn, wq, BS, D, D, Qf, nullptr, nullptr, 0);
    gemm(Xn, wk, BS, D, D, Kf, nullptr, nullptr, 0);
    gemm(Xn, wv, BS, D, D, nullptr, Vb, nullptr, 4);
    gemm(Xn, wg, BS, D, D, nullptr, Gb, nullptr, 2);      // swish gate
    rotary_kernel<<<(BS * 8 * 64) / 256, 256, 0, stream>>>(Qf, Kf, Qb, Kb);
    retention_kernel<<<dim3(S / 128, 8, B), 256, 0, stream>>>(Qb, Kb, Vb, Yw);
    gate_gn_kernel<<<BS, 256, 0, stream>>>(Yw, Gb, gns + l * D, gnb + l * D, GY);
    gemm(GY, wo, BS, D, D, Yr, nullptr, Xcur, 1);         // + residual X
    ln_kernel<<<BS, 256, 0, stream>>>(Yr, Xn, lns, lnb);
    gemm(Xn, w1, BS, F, D, nullptr, H1, nullptr, 3);      // gelu
    gemm(H1, w2, BS, D, F, out, nullptr, Yr, 1);          // + residual Yr
  }
}
